// GaussianVectorQuantizer_14078902796353
// MI455X (gfx1250) — matrix operand order repaired and checked by reading, not compiled, not run on hardware
//
#include <hip/hip_runtime.h>

#define B_ 8
#define N_ 2048
#define D_ 64
#define K_ 8
#define S_ 256
#define TEMP_INV 2.0f      // 1/TEMP, TEMP=0.5
#define EPS_ 1e-10f

#define NT 64              // rows per workgroup
#define BLOCK 128          // 4 waves (wave32)

// output layout (floats): zq | precision_q | prob | log_prob
#define ZQ_OFF   0
#define PQ_OFF   (B_*N_*D_)
#define PROB_OFF (PQ_OFF + 1)
#define LOGP_OFF (PROB_OFF + B_*N_*S_)

// workspace layout
#define WS_CPROB 0                       // 64 f32
#define WS_BS2   64                      // 2048 f32
#define WS_PQ    2112
#define WS_PQCLS 2113
#define WS_B1_BYTE 10240                 // K*S*D bf16, GEMM1 B frag-major
#define WS_B2_BYTE (WS_B1_BYTE + K_*S_*D_*2)  // K*S*D bf16, GEMM2 B frag-major

typedef __attribute__((ext_vector_type(16))) __bf16 v16bf;
typedef __attribute__((ext_vector_type(8)))  float  v8f;
typedef unsigned __attribute__((ext_vector_type(4))) u32x4;
typedef int      __attribute__((ext_vector_type(4))) i32x4;
typedef int      __attribute__((ext_vector_type(8))) i32x8;

#if defined(__has_builtin)
#if __has_builtin(__builtin_amdgcn_tensor_load_to_lds) && __has_builtin(__builtin_amdgcn_s_wait_tensorcnt)
#define HAVE_TDM 1
#endif
#endif

struct U16x16 { uint4 a, b; };

static __device__ __forceinline__ v16bf frag_from(const uint4* p) {
  U16x16 u; u.a = p[0]; u.b = p[1];
  return __builtin_bit_cast(v16bf, u);
}

static __device__ __forceinline__ __bf16 f2bf(float f) {
  unsigned u = __builtin_bit_cast(unsigned, f);
  u += 0x7FFFu + ((u >> 16) & 1u);     // round-to-nearest-even
  unsigned short h = (unsigned short)(u >> 16);
  return __builtin_bit_cast(__bf16, h);
}
// two f32 -> packed bf16 pair in one dword (lo=a, hi=b)
static __device__ __forceinline__ unsigned pack2bf(float a, float b) {
  unsigned ua = __builtin_bit_cast(unsigned, a); ua += 0x7FFFu + ((ua >> 16) & 1u);
  unsigned ub = __builtin_bit_cast(unsigned, b); ub += 0x7FFFu + ((ub >> 16) & 1u);
  return (ua >> 16) | (ub & 0xFFFF0000u);
}
// 8 consecutive f32 (two float4) -> uint4 of packed bf16
static __device__ __forceinline__ uint4 pack8bf(float4 lo, float4 hi) {
  uint4 q;
  q.x = pack2bf(lo.x, lo.y); q.y = pack2bf(lo.z, lo.w);
  q.z = pack2bf(hi.x, hi.y); q.w = pack2bf(hi.z, hi.w);
  return q;
}

// deterministic gumbel(0,1) from a 32-bit index (lowbias32 hash)
static __device__ __forceinline__ float gumbel_of(unsigned idx) {
  unsigned x = idx * 0x9E3779B1u + 0x85EBCA77u;
  x ^= x >> 16; x *= 0x7FEB352Du;
  x ^= x >> 15; x *= 0x846CA68Bu;
  x ^= x >> 16;
  float U = ((float)(x >> 8) + 0.5f) * (1.0f / 16777216.0f);   // (0,1)
  return -__logf(-__logf(U + EPS_) + EPS_);
}

// ---------------------------------------------------------------------------
// Prep: scalars, c_prob (gumbel-softmax over K), codebook norms, and bf16
// codebook reorders into WMMA B-fragment-major layouts.
// ---------------------------------------------------------------------------
__global__ void gvq_prep(const float* __restrict__ c_logits,
                         const float* __restrict__ books,
                         const float* __restrict__ lpq,
                         const float* __restrict__ lpqc,
                         float* __restrict__ ws,
                         float* __restrict__ out) {
  const int t = threadIdx.x;
  const float pq  = 0.5f / fmaxf(1.0f + __expf(lpq[0]),  1e-10f);
  const float pqc = 0.5f / fmaxf(1.0f + __expf(lpqc[0]), 1e-10f);
  if (t == 0) { ws[WS_PQ] = pq; ws[WS_PQCLS] = pqc; out[PQ_OFF] = pq; }

  // c_prob: 64 lanes, groups of 8 (one batch per group), shfl_xor reduce
  if (t < 64) {
    int b = t >> 3, k = t & 7;
    float g = gumbel_of(0xC0FFEE00u + (unsigned)(b * K_ + k));
    float x = (c_logits[b * K_ + k] * pqc + g) * TEMP_INV;
    float mx = x;
    #pragma unroll
    for (int msk = 1; msk < 8; msk <<= 1) mx = fmaxf(mx, __shfl_xor(mx, msk, 32));
    float e = __expf(x - mx), s = e;
    #pragma unroll
    for (int msk = 1; msk < 8; msk <<= 1) s += __shfl_xor(s, msk, 32);
    ws[WS_CPROB + t] = e / s;
  }

  // codebook row norms ||books[k,s]||^2
  for (int idx = t; idx < K_ * S_; idx += blockDim.x) {
    const float* bp = books + (size_t)idx * D_;
    float acc = 0.f;
    #pragma unroll 4
    for (int d = 0; d < D_; ++d) { float v = bp[d]; acc += v * v; }
    ws[WS_BS2 + idx] = acc;
  }

  __bf16* wsB1 = (__bf16*)((char*)ws + WS_B1_BYTE);
  __bf16* wsB2 = (__bf16*)((char*)ws + WS_B2_BYTE);
  const int TOT = K_ * S_ * D_;

  // GEMM1 B (books^T per k): B[K=d][N=s]; frag layout [k][st][kk][lane][i]
  // value = books[k][ s = st*16 + (lane&15) ][ d = kk*32 + (i>>3)*16 + (lane>>4)*8 + (i&7) ]
  for (int e = t; e < TOT; e += blockDim.x) {
    int i = e & 15, lane = (e >> 4) & 31, kk = (e >> 9) & 1, st = (e >> 10) & 15, k = e >> 14;
    int s = st * 16 + (lane & 15);
    int d = kk * 32 + (i >> 3) * 16 + (lane >> 4) * 8 + (i & 7);
    wsB1[e] = f2bf(books[((size_t)k * S_ + s) * D_ + d]);
  }
  // GEMM2 B (books per k): B[K=s][N=d]; frag layout [k][dt][ss][lane][i]
  // value = books[k][ s = ss*32 + (i>>3)*16 + (lane>>4)*8 + (i&7) ][ d = dt*16 + (lane&15) ]
  for (int e = t; e < TOT; e += blockDim.x) {
    int i = e & 15, lane = (e >> 4) & 31, ss = (e >> 9) & 7, dt = (e >> 12) & 3, k = e >> 14;
    int s = ss * 32 + (i >> 3) * 16 + (lane >> 4) * 8 + (i & 7);
    int d = dt * 16 + (lane & 15);
    wsB2[e] = f2bf(books[((size_t)k * S_ + s) * D_ + d]);
  }
}

// ---------------------------------------------------------------------------
// Main fused kernel: per (b, 64-row tile):
//   TDM loads the ze tile to LDS (tensor_load_to_lds, TENSORcnt), then
//   for k: WMMA GEMM1 -> logit_k (regs) -> mix accum (LDS) ->
//          gumbel-softmax (regs + shfl_xor) -> enc bf16 (LDS, c_prob folded)
//          -> WMMA GEMM2 accumulating zq fragments
//   then zq stores + row softmax of mix -> prob / log_prob
// ---------------------------------------------------------------------------
__global__ void __launch_bounds__(BLOCK)
gvq_main(const float* __restrict__ ze,
         const float* __restrict__ ws,
         float* __restrict__ out) {
  extern __shared__ char smem[];
  float*  mixS    = (float*)smem;                                  // NT*S f32 (64KB)
  __bf16* encS    = (__bf16*)(smem + NT * S_ * 4);                 // NT*S bf16 (32KB)
  float*  zeStage = (float*)(smem + NT * S_ * 4 + NT * S_ * 2);    // NT*D f32 (16KB)
  float*  zn2S    = (float*)(smem + NT * S_ * 4 + NT * S_ * 2 + NT * D_ * 4); // NT f32

  const int tid  = threadIdx.x;
  const int lane = tid & 31;
  const int wave = tid >> 5;
  const int m    = lane & 15;       // column-in-tile / A-row-in-block
  const int half = lane >> 4;       // K half-group
  const int rowbase = wave * 16;    // this wave's 16-row block

  const int wg = blockIdx.x;
  const int b  = wg >> 5;           // 32 row-tiles per batch
  const int n0 = (wg & 31) * NT;

  const __bf16* wsB1 = (const __bf16*)((const char*)ws + WS_B1_BYTE);
  const __bf16* wsB2 = (const __bf16*)((const char*)ws + WS_B2_BYTE);

  // ---- stage ze tile (64 rows x 64 f32) into LDS ----
#if HAVE_TDM
  if (wave == 0) {
    // D# group0: count=1 | lds_addr | global_addr[56:0] | type=2
    unsigned long long ga = (unsigned long long)(const void*)(ze + ((size_t)(b * N_ + n0)) * D_);
    unsigned lds_addr = (unsigned)(unsigned long long)(void*)zeStage; // LDS aperture: low 32 bits = LDS byte offset
    u32x4 g0;
    g0[0] = 1u;                                   // count=1, is_restore=0, gather off
    g0[1] = lds_addr;
    g0[2] = (unsigned)(ga & 0xFFFFFFFFull);
    g0[3] = (unsigned)((ga >> 32) & 0x01FFFFFFull) | (2u << 30);   // type=2 ("image")
    // D# group1: data_size=4B, tensor 64x2048, tile 64x64, dim0_stride=64
    i32x8 g1;
    g1[0] = 0x00020000;          // workgroup_mask=0, data_size=2 (4B)
    g1[1] = (int)(64u << 16);    // tensor_dim0 low16 = 64 (atomic_barrier_addr=0)
    g1[2] = (int)(2048u << 16);  // tensor_dim0 hi=0, tensor_dim1 low16 = 2048
    g1[3] = (int)(64u << 16);    // tensor_dim1 hi=0, tile_dim0 = 64
    g1[4] = 64;                  // tile_dim1 = 64, tile_dim2 = 0
    g1[5] = 64;                  // tensor_dim0_stride low32 = 64
    g1[6] = 0;                   // dim0_stride hi, dim1_stride low
    g1[7] = 0;
    i32x4 z4; z4[0] = 0; z4[1] = 0; z4[2] = 0; z4[3] = 0;
#if __clang_major__ >= 23
    i32x8 z8; for (int i = 0; i < 8; ++i) z8[i] = 0;
    __builtin_amdgcn_tensor_load_to_lds(g0, g1, z4, z4, z8, 0);
#else
    __builtin_amdgcn_tensor_load_to_lds(g0, g1, z4, z4, 0);
#endif
    __builtin_amdgcn_s_wait_tensorcnt(0);
  }
#else
  for (int i = tid; i < NT * D_ / 4; i += BLOCK)
    ((float4*)zeStage)[i] = ((const float4*)(ze + ((size_t)(b * N_ + n0)) * D_))[i];
#endif
  for (int i = tid; i < NT * S_; i += BLOCK) mixS[i] = 0.f;   // zero mix while TDM flies
  __syncthreads();

  // row norms ||ze[n]||^2 from the LDS tile
  for (int r = tid; r < NT; r += BLOCK) {
    const float4* zp = (const float4*)(zeStage + r * D_);
    float acc = 0.f;
    #pragma unroll
    for (int d4 = 0; d4 < D_ / 4; ++d4) {
      float4 v = zp[d4];
      acc += v.x * v.x + v.y * v.y + v.z * v.z + v.w * v.w;
    }
    zn2S[r] = acc;
  }
  __syncthreads();

  // A fragments of ze (documented 16-bit A layout: lane holds row m,
  // halves i: K = kk*32 + (i<8?0:16) + half*8 + (i&7)); packed f32->bf16
  v16bf afrag[2];
  #pragma unroll
  for (int kk = 0; kk < 2; ++kk) {
    const float4* c0 = (const float4*)(zeStage + (rowbase + m) * D_ + kk * 32 + half * 8);
    const float4* c1 = (const float4*)(zeStage + (rowbase + m) * D_ + kk * 32 + 16 + half * 8);
    U16x16 u;
    u.a = pack8bf(c0[0], c0[1]);
    u.b = pack8bf(c1[0], c1[1]);
    afrag[kk] = __builtin_bit_cast(v16bf, u);
  }
  float zn2r[8];
  #pragma unroll
  for (int j = 0; j < 8; ++j) zn2r[j] = zn2S[rowbase + j + 8 * half];

  const float pq = ws[WS_PQ];

  v8f zqacc[4];
  #pragma unroll
  for (int dt = 0; dt < 4; ++dt)
    #pragma unroll
    for (int j = 0; j < 8; ++j) zqacc[dt][j] = 0.f;

  for (int k = 0; k < K_; ++k) {
    const float cpk = ws[WS_CPROB + b * K_ + k];
    if (k + 1 < K_) {
      __builtin_prefetch(wsB1 + (size_t)(k + 1) * 16 * 2 * 32 * 16, 0, 1);
      __builtin_prefetch(wsB2 + (size_t)(k + 1) * 4 * 8 * 32 * 16, 0, 1);
    }

    // ---- GEMM1: logit_k tiles for this wave's 16 rows x 256 cols ----
    v8f lf[16];
    #pragma unroll
    for (int st = 0; st < 16; ++st) {
      v8f c;
      #pragma unroll
      for (int j = 0; j < 8; ++j) c[j] = 0.f;
      #pragma unroll
      for (int kk = 0; kk < 2; ++kk) {
        const uint4* bp = (const uint4*)(wsB1 + ((size_t)((k * 16 + st) * 2 + kk) * 32 + lane) * 16);
        v16bf bf = frag_from(bp);
        c = __builtin_amdgcn_wmma_f32_16x16x32_bf16(false, afrag[kk], false, bf,
                                                    (short)0, c, false, false);
      }
      float bsv = pq * ws[WS_BS2 + k * S_ + st * 16 + m];
      #pragma unroll
      for (int j = 0; j < 8; ++j)
        lf[st][j] = 2.0f * pq * c[j] - pq * zn2r[j] - bsv;   // -dist*precision_q
    }

    // ---- mixture logits accumulation (wave owns its rows; no races) ----
    #pragma unroll
    for (int st = 0; st < 16; ++st)
      #pragma unroll
      for (int j = 0; j < 8; ++j)
        mixS[(rowbase + j + 8 * half) * S_ + st * 16 + m] += cpk * lf[st][j];

    // ---- gumbel noise + 1/TEMP, row max (regs then 16-lane shfl_xor) ----
    float mx[8];
    #pragma unroll
    for (int j = 0; j < 8; ++j) mx[j] = -3.0e38f;
    unsigned base =
        ((unsigned)(b * K_ + k) * N_ + (unsigned)(n0 + rowbase + 8 * half)) * S_;
    #pragma unroll
    for (int st = 0; st < 16; ++st)
      #pragma unroll
      for (int j = 0; j < 8; ++j) {
        unsigned id = base + (unsigned)j * S_ + (unsigned)(st * 16 + m);
        float g = gumbel_of(id);
        float tv = (lf[st][j] + g) * TEMP_INV;
        lf[st][j] = tv;
        mx[j] = fmaxf(mx[j], tv);
      }
    #pragma unroll
    for (int j = 0; j < 8; ++j)
      #pragma unroll
      for (int msk = 1; msk < 16; msk <<= 1)
        mx[j] = fmaxf(mx[j], __shfl_xor(mx[j], msk, 32));

    float sm[8];
    #pragma unroll
    for (int j = 0; j < 8; ++j) sm[j] = 0.f;
    #pragma unroll
    for (int st = 0; st < 16; ++st)
      #pragma unroll
      for (int j = 0; j < 8; ++j) {
        float e = __expf(lf[st][j] - mx[j]);
        lf[st][j] = e; sm[j] += e;
      }
    #pragma unroll
    for (int j = 0; j < 8; ++j)
      #pragma unroll
      for (int msk = 1; msk < 16; msk <<= 1)
        sm[j] += __shfl_xor(sm[j], msk, 32);

    float scl[8];
    #pragma unroll
    for (int j = 0; j < 8; ++j) scl[j] = cpk / sm[j];   // fold c_prob into enc
    #pragma unroll
    for (int st = 0; st < 16; ++st)
      #pragma unroll
      for (int j = 0; j < 8; ++j)
        encS[(rowbase + j + 8 * half) * S_ + st * 16 + m] = f2bf(lf[st][j] * scl[j]);
    __syncthreads();

    // ---- GEMM2: zq += enc_k @ books_k ----
    #pragma unroll
    for (int dt = 0; dt < 4; ++dt) {
      v8f c = zqacc[dt];
      #pragma unroll
      for (int ss = 0; ss < 8; ++ss) {
        const uint4* p0 = (const uint4*)(encS + (rowbase + m) * S_ + ss * 32 + half * 8);
        const uint4* p1 = (const uint4*)(encS + (rowbase + m) * S_ + ss * 32 + 16 + half * 8);
        U16x16 ua; ua.a = *p0; ua.b = *p1;
        v16bf ea = __builtin_bit_cast(v16bf, ua);
        const uint4* bp = (const uint4*)(wsB2 + ((size_t)((k * 4 + dt) * 8 + ss) * 32 + lane) * 16);
        v16bf bb = frag_from(bp);
        c = __builtin_amdgcn_wmma_f32_16x16x32_bf16(false, ea, false, bb,
                                                    (short)0, c, false, false);
      }
      zqacc[dt] = c;
    }
    __syncthreads();
  }

  // ---- zq stores ----
  #pragma unroll
  for (int dt = 0; dt < 4; ++dt)
    #pragma unroll
    for (int j = 0; j < 8; ++j) {
      int row = n0 + rowbase + j + 8 * half;
      int col = dt * 16 + m;
      __builtin_nontemporal_store(zqacc[dt][j], out + ((size_t)(b * N_ + row)) * D_ + col);
    }

  // ---- final row softmax of mixture logits -> prob / log_prob ----
  __syncthreads();
  {
    int r  = tid >> 1;              // row (each wave reads only rows it wrote)
    int sh = (tid & 1) * 128;       // half of the 256 columns
    const float4* rowq = (const float4*)(mixS + r * S_ + sh);
    float mx = -3.0e38f;
    #pragma unroll 4
    for (int s4 = 0; s4 < 32; ++s4) {
      float4 v = rowq[s4];
      mx = fmaxf(mx, fmaxf(fmaxf(v.x, v.y), fmaxf(v.z, v.w)));
    }
    mx = fmaxf(mx, __shfl_xor(mx, 1, 32));
    float sum = 0.f;
    #pragma unroll 4
    for (int s4 = 0; s4 < 32; ++s4) {
      float4 v = rowq[s4];
      sum += __expf(v.x - mx) + __expf(v.y - mx) + __expf(v.z - mx) + __expf(v.w - mx);
    }
    sum += __shfl_xor(sum, 1, 32);
    float inv = 1.0f / sum;
    float lse = __logf(sum);
    size_t obase = ((size_t)(b * N_ + n0 + r)) * S_ + sh;
    float* pp = out + PROB_OFF + obase;
    float* lp = out + LOGP_OFF + obase;
    #pragma unroll 2
    for (int s4 = 0; s4 < 32; ++s4) {
      float4 v = rowq[s4];
      __builtin_nontemporal_store(__expf(v.x - mx) * inv, pp + s4 * 4 + 0);
      __builtin_nontemporal_store(__expf(v.y - mx) * inv, pp + s4 * 4 + 1);
      __builtin_nontemporal_store(__expf(v.z - mx) * inv, pp + s4 * 4 + 2);
      __builtin_nontemporal_store(__expf(v.w - mx) * inv, pp + s4 * 4 + 3);
      __builtin_nontemporal_store(v.x - mx - lse, lp + s4 * 4 + 0);
      __builtin_nontemporal_store(v.y - mx - lse, lp + s4 * 4 + 1);
      __builtin_nontemporal_store(v.z - mx - lse, lp + s4 * 4 + 2);
      __builtin_nontemporal_store(v.w - mx - lse, lp + s4 * 4 + 3);
    }
  }
}

extern "C" void kernel_launch(void* const* d_in, const int* in_sizes, int n_in,
                              void* d_out, int out_size, void* d_ws, size_t ws_size,
                              hipStream_t stream) {
  (void)in_sizes; (void)n_in; (void)out_size; (void)ws_size;
  const float* ze       = (const float*)d_in[0];
  const float* c_logits = (const float*)d_in[1];
  const float* books    = (const float*)d_in[2];
  const float* lpq      = (const float*)d_in[3];
  const float* lpqc     = (const float*)d_in[4];
  float* out = (float*)d_out;
  float* ws  = (float*)d_ws;

  gvq_prep<<<dim3(1), dim3(256), 0, stream>>>(c_logits, books, lpq, lpqc, ws, out);

  const size_t smem_bytes = (size_t)NT * S_ * 4   // mix f32
                          + (size_t)NT * S_ * 2   // enc bf16
                          + (size_t)NT * D_ * 4   // ze f32 stage (TDM dest)
                          + (size_t)NT * 4;       // zn2
  gvq_main<<<dim3(B_ * (N_ / NT)), dim3(BLOCK), smem_bytes, stream>>>(ze, ws, out);
}
